// BhattacharyyaCoeff_24068996726831
// MI455X (gfx1250) — compile-verified, matrix operand order checked by reading, not run on hardware
//
#include <hip/hip_runtime.h>
#include <hip/hip_bf16.h>

typedef __attribute__((ext_vector_type(16))) __bf16 v16bf;
typedef __attribute__((ext_vector_type(8)))  __bf16 v8bf;
typedef __attribute__((ext_vector_type(8)))  float  v8f;

#define N_    16
#define C_    256
#define M_    63
#define MO_   56
#define CGRP  16            // channels per staging iteration
#define NCG   (C_ / CGRP)   // 16
#define ORWS  14            // output rows per workgroup
#define ROWSP 24            // padded input rows: 8 waves x 3 rows (>= 14+7)
#define JT    16            // output-column tile
#define COLS  24            // staged columns (16 + 8 halo)

__device__ __forceinline__ __bf16 f2bf(float f) {
  // round-to-nearest-even f32 -> bf16 without relying on __bf16 arithmetic
  unsigned u = __builtin_bit_cast(unsigned, f);
  unsigned r = u + 0x7fffu + ((u >> 16) & 1u);
  return __builtin_bit_cast(__bf16, (unsigned short)(r >> 16));
}

__global__ __launch_bounds__(256)
void BhattacharyyaCoeff_24068996726831_kernel(const float* __restrict__ z,
                                              const float* __restrict__ x,
                                              const float* __restrict__ w,
                                              float* __restrict__ out) {
  // w*sqrt(z[n]) for all channels, layout [ch][p][q]; rows of 8 bf16, 16B-aligned
  __shared__ __align__(16) __bf16 sZ[C_ * 64];                 // 32 KB
  // sqrt(x[n]) slab for one channel group: [ch][row][col]
  __shared__ __align__(16) __bf16 sX[CGRP][ROWSP][COLS];       // 18 KB
  // per-input-row GEMM results D_r[j, p] (exclusive ownership, no atomics)
  __shared__ float sD[ROWSP][JT][8];                           // 12 KB
  // 16-byte zero block for branchless B-fragment padding (p >= 8 lanes)
  __shared__ __align__(16) __bf16 sZero[8];

  const int n  = blockIdx.x;
  const int jt = blockIdx.y;
  const int it = blockIdx.z;
  const int j0 = jt * JT;
  const int i0 = it * ORWS;

  const int tid  = threadIdx.x;
  const int wave = tid >> 5;
  const int lane = tid & 31;
  const int m    = lane & 15;  // A-row (output column j) / B-column (kernel row p)
  const int h    = lane >> 4;  // K-half select per the 16-bit WMMA lane layout

  if (tid < 8) sZero[tid] = f2bf(0.0f);

  // Stage w*sqrt(z[n]) once: 16384 elements, 64 per thread (uniform trip count).
  for (int idx = tid; idx < C_ * 64; idx += 256) {
    int ch = idx >> 6;
    sZ[idx] = f2bf(w[ch] * sqrtf(z[(n * C_ + ch) * 64 + (idx & 63)]));
  }
  __syncthreads();

  v8f acc[3] = {};             // 3 independent D_r accumulator chains per wave
  const int r0 = wave * 3;     // this wave's rows: r0, r0+1, r0+2 (no guards)

  for (int cg = 0; cg < NCG; ++cg) {
    __syncthreads();           // protect sX from previous iteration's readers
    // Stage sqrt(x) slab: 16 ch x 24 rows x 24 cols = 9216 = 36*256 (uniform)
    for (int idx = tid; idx < CGRP * ROWSP * COLS; idx += 256) {
      int ch  = idx / (ROWSP * COLS);
      int rem = idx - ch * (ROWSP * COLS);
      int row = rem / COLS;
      int col = rem - row * COLS;
      int gr  = i0 + row;
      int gc  = j0 + col;
      float v = 0.0f;
      if (gr < M_ && gc < M_)
        v = x[((n * C_ + cg * CGRP + ch) * M_ + gr) * M_ + gc];
      sX[ch][row][col] = f2bf(sqrtf(v));
    }
    __syncthreads();

    // Prefetch next channel group's x slab into cache (global_prefetch_b8).
    if (cg + 1 < NCG) {
      for (int idx = tid; idx < CGRP * ROWSP; idx += 256) {
        int ch  = idx / ROWSP;
        int row = idx - ch * ROWSP;
        int gr  = i0 + row; if (gr > M_ - 1) gr = M_ - 1;
        __builtin_prefetch(&x[((n * C_ + (cg + 1) * CGRP + ch) * M_ + gr) * M_ + j0], 0, 1);
      }
    }

    // B fragments (z side), branchless padding for p >= 8 via sZero select.
    v16bf bfrag[4];
    #pragma unroll
    for (int cc = 0; cc < 4; ++cc) {
      const __bf16* plo = (m < 8) ? &sZ[((cg * CGRP + 4 * cc + h)     << 6) + m * 8] : sZero;
      const __bf16* phi = (m < 8) ? &sZ[((cg * CGRP + 4 * cc + 2 + h) << 6) + m * 8] : sZero;
      bfrag[cc] = __builtin_shufflevector(*(const v8bf*)plo, *(const v8bf*)phi,
                    0, 1, 2, 3, 4, 5, 6, 7, 8, 9, 10, 11, 12, 13, 14, 15);
    }

    // Chunk-outer / row-inner: 3 independent WMMA chains hide D->C latency.
    #pragma unroll
    for (int cc = 0; cc < 4; ++cc) {
      #pragma unroll
      for (int rr = 0; rr < 3; ++rr) {
        const __bf16* ra = &sX[4 * cc + h][r0 + rr][m];      // sliding window
        const __bf16* rb = &sX[4 * cc + 2 + h][r0 + rr][m];
        v16bf af;
        #pragma unroll
        for (int e = 0; e < 8; ++e) { af[e] = ra[e]; af[8 + e] = rb[e]; }
        acc[rr] = __builtin_amdgcn_wmma_f32_16x16x32_bf16(
            false, af, false, bfrag[cc], (short)0, acc[rr], false, false);
      }
    }
  }

  // Write D_r tiles to LDS: lane (p=m, h) owns D[j = v+8h, p] for its 3 rows.
  if (m < 8) {
    #pragma unroll
    for (int rr = 0; rr < 3; ++rr) {
      #pragma unroll
      for (int v = 0; v < 8; ++v)
        sD[r0 + rr][v + 8 * h][m] = acc[rr][v];
    }
  }
  __syncthreads();

  // out[i, j] = (1/64) * sum_p D_{i+p}[j, p]   (rows 21..23 never read)
  if (tid < ORWS * JT) {
    int il = tid >> 4, jl = tid & 15;
    int j = j0 + jl;
    if (j < MO_) {
      float s = 0.0f;
      #pragma unroll
      for (int p = 0; p < 8; ++p) s += sD[il + p][jl][p];
      out[(n * MO_ + (i0 + il)) * MO_ + j] = s * (1.0f / 64.0f);
    }
  }
}

extern "C" void kernel_launch(void* const* d_in, const int* in_sizes, int n_in,
                              void* d_out, int out_size, void* d_ws, size_t ws_size,
                              hipStream_t stream) {
  const float* z = (const float*)d_in[0];   // (16, 256, 8, 8)   f32
  const float* x = (const float*)d_in[1];   // (16, 256, 63, 63) f32
  const float* w = (const float*)d_in[2];   // (256,)            f32
  float* out = (float*)d_out;               // (16, 1, 56, 56)   f32

  dim3 grid(N_, 4, 4);                      // (n, j-tile, i-tile)
  hipLaunchKernelGGL(BhattacharyyaCoeff_24068996726831_kernel,
                     grid, dim3(256), 0, stream, z, x, w, out);
}
